// Model_77773267796096
// MI455X (gfx1250) — matrix operand order repaired and checked by reading, not compiled, not run on hardware
//
#include <hip/hip_runtime.h>

// ---------------------------------------------------------------------------
// RITS recurrence for MI455X (gfx1250, wave32, WMMA).
//   B=8192, T=80, F=12, H=128, gates=512.
// Gate GEMMs via v_wmma_f32_16x16x32_f16 with weights pre-swizzled to f16
// fragment layout in LDS (contiguous k = 16h + s per lane half). All elementwise /
// K=12 GEMMs / activations in f32 VALU. ~126 MB HBM traffic total.
// ---------------------------------------------------------------------------

typedef __attribute__((ext_vector_type(16))) _Float16 v16h;
typedef __attribute__((ext_vector_type(8)))  float    v8f;

#define B_TOTAL 8192
#define T_SEQ   80
#define F_IN    12
#define H_RNN   128
#define G4H     512      // 4*H
#define TILE_B  16
#define NTILES  32       // G4H / 16
#define KCH     4        // H / 32

// LDS carve sizes (elements)
#define N_WHH_FRAG (NTILES*KCH*32*16)   // 65536 halves
#define N_WIH_FRAG (NTILES*32*16)       // 16384 halves
#define N_AFRAG_H  (KCH*32*16)          // 2048  halves
#define N_AFRAG_I  (32*16)              // 512   halves
// bytes: halves = (65536+16384+2048+512)*2 = 168960
//        floats = 8192+2048+2048+1536+128+1536+16+512+128+4*192+4 = 16916 -> 67664
#define LDS_BYTES (168960 + 67664)      // 236624 B  (< 320 KB/WGP)

__device__ __forceinline__ float sig_(float x) { return 1.f / (1.f + __expf(-x)); }

__global__ void __launch_bounds__(256, 1)
rits_kernel(const float* __restrict__ values,  // [B,T,F]
            const float* __restrict__ masks,   // [B,T,F]
            const float* __restrict__ deltas,  // [B,T,F]
            const float* __restrict__ W_td,    // [H,F]
            const float* __restrict__ b_td,    // [H]
            const float* __restrict__ W_reg,   // [F,H]
            const float* __restrict__ b_reg,   // [F]
            const float* __restrict__ W_ih,    // [4H,2F]
            const float* __restrict__ W_hh,    // [4H,H]
            const float* __restrict__ b_ih,    // [4H]
            const float* __restrict__ b_hh,    // [4H]
            const float* __restrict__ W_out,   // [1,H]
            const float* __restrict__ b_out,   // [1]
            float* __restrict__ out_yh,        // [B]
            float* __restrict__ out_imp,       // [B,T,F]
            float* __restrict__ g_num,         // [T]
            float* __restrict__ g_den)         // [T]
{
    extern __shared__ char smem_raw[];
    _Float16* Whh_frag = (_Float16*)smem_raw;               // pre-swizzled B frags
    _Float16* Wih_frag = Whh_frag + N_WHH_FRAG;
    _Float16* AfragH   = Wih_frag + N_WIH_FRAG;             // per-step A frags (h)
    _Float16* AfragI   = AfragH   + N_AFRAG_H;              // per-step A frag (inp)
    float* gatesS = (float*)(AfragI + N_AFRAG_I);           // [16][512]
    float* hbuf   = gatesS + TILE_B*G4H;                    // [16][128]
    float* cbuf   = hbuf   + TILE_B*H_RNN;                  // [16][128]
    float* WtdS   = cbuf   + TILE_B*H_RNN;                  // [128][12]
    float* btdS   = WtdS   + H_RNN*F_IN;                    // [128]
    float* WregS  = btdS   + H_RNN;                         // [12][128]
    float* bregS  = WregS  + F_IN*H_RNN;                    // [16]
    float* bsumS  = bregS  + 16;                            // [512] b_ih+b_hh
    float* WoutS  = bsumS  + G4H;                           // [128]
    float* xsS    = WoutS  + H_RNN;                         // [16*12]
    float* msS    = xsS + TILE_B*F_IN;
    float* dlS    = msS + TILE_B*F_IN;
    float* xcS    = dlS + TILE_B*F_IN;
    float* redS   = xcS + TILE_B*F_IN;                      // [2] num/den partials
    float* boutS  = redS + 2;                               // [1]

    const int tid  = threadIdx.x;
    const int lane = tid & 31;
    const int wv   = tid >> 5;
    const long b0  = (long)blockIdx.x * TILE_B;

    // ---- one-time: small weights + biases to LDS, zero state ----
    for (int i = tid; i < G4H; i += 256) bsumS[i] = b_ih[i] + b_hh[i];
    for (int i = tid; i < H_RNN*F_IN; i += 256) { WtdS[i] = W_td[i]; WregS[i] = W_reg[i]; }
    for (int i = tid; i < H_RNN; i += 256) { btdS[i] = b_td[i]; WoutS[i] = W_out[i]; }
    if (tid < F_IN) bregS[tid] = b_reg[tid];
    if (tid == 0)  boutS[0] = b_out[0];
    for (int i = tid; i < TILE_B*H_RNN; i += 256) { hbuf[i] = 0.f; cbuf[i] = 0.f; }

    // ---- one-time: W_hh -> f16 B-fragments in LDS ----
    // B frags (32x16 KxN), this code's contiguous k order (not AMD's table order): lanes 0-15 hold K=0..15, lanes 16-31 hold K=16..31;
    // element i within a lane's v16h is K = (lane>=16)*16 + i, N = lane&15.
    for (int e = tid; e < N_WHH_FRAG; e += 256) {
        int i  = e & 15;
        int r  = e >> 4;
        int ln = r & 31;
        int f  = r >> 5;                 // frag = tile*KCH + kc
        int kc = f & (KCH - 1), tl = f >> 2;
        int n  = ln & 15;
        int k  = kc*32 + ((ln >> 4) << 4) + i;
        Whh_frag[e] = (_Float16)W_hh[(tl*16 + n)*H_RNN + k];
    }
    // W_ih (K=24 zero-padded to 32) -> f16 B-fragments
    for (int e = tid; e < N_WIH_FRAG; e += 256) {
        int i  = e & 15;
        int r  = e >> 4;
        int ln = r & 31;
        int tl = r >> 5;
        int n  = ln & 15;
        int k  = ((ln >> 4) << 4) + i;
        Wih_frag[e] = (k < 2*F_IN) ? (_Float16)W_ih[(tl*16 + n)*(2*F_IN) + k]
                                   : (_Float16)0.f;
    }
    __syncthreads();

    // ================= time loop (sequential recurrence) =================
    for (int t = 0; t < T_SEQ; ++t) {
        // -- stage x, m, d for this step; reset loss partials --
        if (tid < TILE_B*F_IN) {
            int row = tid / F_IN, f = tid % F_IN;
            long gi = ((b0 + row)*T_SEQ + t)*F_IN + f;
            xsS[tid] = values[gi];
            msS[tid] = masks[gi];
            dlS[tid] = deltas[gi];
        }
        if (tid == 0) { redS[0] = 0.f; redS[1] = 0.f; }
        __syncthreads();

        // -- temporal decay: h *= exp(-relu(d @ W_td^T + b_td)) --
        for (int e = tid; e < TILE_B*H_RNN; e += 256) {
            int row = e >> 7, j = e & 127;
            float acc = btdS[j];
#pragma unroll
            for (int k = 0; k < F_IN; ++k)
                acc = fmaf(dlS[row*F_IN + k], WtdS[j*F_IN + k], acc);
            hbuf[e] *= __expf(-fmaxf(acc, 0.f));
        }
        __syncthreads();

        // -- regression x_h = h @ W_reg^T + b_reg; x_c; loss partials; imputation --
        if (tid < TILE_B*F_IN) {
            int row = tid / F_IN, f = tid % F_IN;
            float acc = bregS[f];
#pragma unroll 8
            for (int j = 0; j < H_RNN; ++j)
                acc = fmaf(hbuf[row*H_RNN + j], WregS[f*H_RNN + j], acc);
            float xv = xsS[tid], mv = msS[tid];
            float xcv = mv*xv + (1.f - mv)*acc;
            xcS[tid] = xcv;
            out_imp[((b0 + row)*T_SEQ + t)*F_IN + f] = xcv;
            atomicAdd(&redS[0], fabsf(xv - acc) * mv);   // ds_add_f32
            atomicAdd(&redS[1], mv);
        }
        __syncthreads();

        // -- build f16 A-fragments (decayed h: 4 K-chunks; inp=[x_c,m] padded to K=32) --
        // A frags (16x32 MxK), this code's contiguous k order (same as B): M = lane&15; elem i -> K = (lane>=16)*16 + i
        for (int e = tid; e < N_AFRAG_H; e += 256) {
            int i  = e & 15;
            int r  = e >> 4;
            int ln = r & 31;
            int kc = r >> 5;
            int m  = ln & 15;
            int k  = kc*32 + ((ln >> 4) << 4) + i;
            AfragH[e] = (_Float16)hbuf[m*H_RNN + k];
        }
        for (int e = tid; e < N_AFRAG_I; e += 256) {
            int i  = e & 15;
            int ln = e >> 4;
            int m  = ln & 15;
            int k  = ((ln >> 4) << 4) + i;
            float v = (k < F_IN)   ? xcS[m*F_IN + k]
                    : (k < 2*F_IN) ? msS[m*F_IN + (k - F_IN)]
                    : 0.f;
            AfragI[e] = (_Float16)v;
        }
        __syncthreads();

        // -- gates = inp@W_ih^T + h@W_hh^T + (b_ih+b_hh) via WMMA --
        // wave wv owns N-tiles [wv*4, wv*4+4); 5 WMMAs each (K=32 inp + 4x K=32 h)
        {
            v16h aI = *(const v16h*)(AfragI + lane*16);
            v16h aH[KCH];
#pragma unroll
            for (int kc = 0; kc < KCH; ++kc)
                aH[kc] = *(const v16h*)(AfragH + (kc*32 + lane)*16);
#pragma unroll
            for (int nt = 0; nt < 4; ++nt) {
                const int tile = wv*4 + nt;
                v8f acc = {};
                v16h bI = *(const v16h*)(Wih_frag + (tile*32 + lane)*16);
                acc = __builtin_amdgcn_wmma_f32_16x16x32_f16(
                        false, aI, false, bI, (short)0, acc, false, false);
#pragma unroll
                for (int kc = 0; kc < KCH; ++kc) {
                    v16h bH = *(const v16h*)(Whh_frag + ((tile*KCH + kc)*32 + lane)*16);
                    acc = __builtin_amdgcn_wmma_f32_16x16x32_f16(
                            false, aH[kc], false, bH, (short)0, acc, false, false);
                }
                // D layout: N = lane&15; VGPR r -> M = r + 8*(lane>=16)
                const int n     = lane & 15;
                const int mbase = (lane >> 4) << 3;
                const float bias = bsumS[tile*16 + n];
#pragma unroll
                for (int r = 0; r < 8; ++r)
                    gatesS[(mbase + r)*G4H + tile*16 + n] = acc[r] + bias;
            }
        }
        __syncthreads();

        // -- LSTM cell update (torch gate order i,f,g,o) --
        for (int e = tid; e < TILE_B*H_RNN; e += 256) {
            int row = e >> 7, j = e & 127;
            const float* gr = gatesS + row*G4H;
            float ig = sig_(gr[j]);
            float fg = sig_(gr[H_RNN + j]);
            float gg = tanhf(gr[2*H_RNN + j]);
            float og = sig_(gr[3*H_RNN + j]);
            float c  = fg * cbuf[e] + ig * gg;
            cbuf[e]  = c;
            hbuf[e]  = og * tanhf(c);
        }
        if (tid == 0) {            // fold this block's loss partials into globals
            atomicAdd(&g_num[t], redS[0]);
            atomicAdd(&g_den[t], redS[1]);
        }
        __syncthreads();
    }

    // ---- y_h = sigmoid(h @ W_out^T + b_out) ----
    if (tid < TILE_B) {
        float acc = boutS[0];
#pragma unroll 8
        for (int j = 0; j < H_RNN; ++j)
            acc = fmaf(hbuf[tid*H_RNN + j], WoutS[j], acc);
        out_yh[b0 + tid] = sig_(acc);
    }
}

__global__ void rits_zero_kernel(float* p, int n) {
    int i = blockIdx.x * blockDim.x + threadIdx.x;
    if (i < n) p[i] = 0.f;
}

__global__ void rits_loss_kernel(const float* __restrict__ num,
                                 const float* __restrict__ den,
                                 float* __restrict__ out_loss) {
    if (threadIdx.x == 0) {
        float s = 0.f;
        for (int t = 0; t < T_SEQ; ++t) s += num[t] / (den[t] + 1e-5f);
        out_loss[0] = s / (float)T_SEQ;
    }
}

extern "C" void kernel_launch(void* const* d_in, const int* in_sizes, int n_in,
                              void* d_out, int out_size, void* d_ws, size_t ws_size,
                              hipStream_t stream) {
    (void)in_sizes; (void)n_in; (void)out_size; (void)ws_size;
    const float* values = (const float*)d_in[0];
    const float* masks_ = (const float*)d_in[1];
    const float* deltas = (const float*)d_in[2];
    const float* W_td   = (const float*)d_in[3];
    const float* b_td   = (const float*)d_in[4];
    const float* W_reg  = (const float*)d_in[5];
    const float* b_reg  = (const float*)d_in[6];
    const float* W_ih   = (const float*)d_in[7];
    const float* W_hh   = (const float*)d_in[8];
    const float* b_ih   = (const float*)d_in[9];
    const float* b_hh   = (const float*)d_in[10];
    const float* W_out  = (const float*)d_in[11];
    const float* b_out  = (const float*)d_in[12];

    float* out      = (float*)d_out;          // [0]=loss, [1..B]=y_h, rest=imputations
    float* out_yh   = out + 1;
    float* out_imp  = out + 1 + B_TOTAL;
    float* g_num    = (float*)d_ws;           // [T] loss numerators
    float* g_den    = g_num + T_SEQ;          // [T] loss denominators

    rits_zero_kernel<<<1, 256, 0, stream>>>(g_num, 2*T_SEQ);
    rits_kernel<<<B_TOTAL/TILE_B, 256, LDS_BYTES, stream>>>(
        values, masks_, deltas, W_td, b_td, W_reg, b_reg,
        W_ih, W_hh, b_ih, b_hh, W_out, b_out,
        out_yh, out_imp, g_num, g_den);
    rits_loss_kernel<<<1, 32, 0, stream>>>(g_num, g_den, out);
}
